// MPSOptimizedAttention_33535104647345
// MI455X (gfx1250) — matrix operand order repaired and checked by reading, not compiled, not run on hardware
//
#include <hip/hip_runtime.h>

typedef __attribute__((ext_vector_type(16))) __bf16 v16bf;
typedef __attribute__((ext_vector_type(8)))  float  v8f;
typedef unsigned short u16;
typedef unsigned int   u32;
typedef unsigned long long u64;

union BFrag { v16bf v; uint4 q[2]; u16 u[16]; };
union F8    { v8f v; float f[8]; };

__device__ __forceinline__ u16 f2bf(float x) {
  u32 u = __float_as_uint(x);
  u32 r = (u + 0x7fffu + ((u >> 16) & 1u)) >> 16;
  return (u16)r;
}

__device__ __forceinline__ v8f wmma_bf16(const BFrag& a, const BFrag& b, v8f c) {
  return __builtin_amdgcn_wmma_f32_16x16x32_bf16(false, a.v, false, b.v, (short)0, c, false, false);
}

// ---------------------------------------------------------------------------
// CDNA5 async global->LDS DMA (ASYNCcnt-tracked), per-lane 16B.
// LDS offset = low 32 bits of the generic pointer (aperture puts the LDS
// byte offset in addr[31:0]).
// ---------------------------------------------------------------------------
__device__ __forceinline__ void async_copy_b128(void* lds, const void* g) {
  const u32 l = (u32)(size_t)lds;
  const u64 a = (u64)(size_t)g;
  asm volatile("global_load_async_to_lds_b128 %0, %1, off"
               :: "v"(l), "v"(a)
               : "memory");
}

template <int N>
__device__ __forceinline__ void wait_asynccnt() {
  asm volatile("s_wait_asynccnt %0" :: "n"(N) : "memory");
}

// ---------------------------------------------------------------------------
// fp32 -> bf16 elementwise conversion
// ---------------------------------------------------------------------------
__global__ void cvt_bf16_kernel(const float* __restrict__ x, u16* __restrict__ y, int n) {
  int i = blockIdx.x * 256 + threadIdx.x;
  if (i < n) y[i] = f2bf(x[i]);
}

// ---------------------------------------------------------------------------
// GEMM: C[M,N] = A[M,K](bf16) * Bw[N,K]^T(bf16) + bias[N], fp32 out.
// Block: 256 threads (8 waves). Tile: 128x128, BK=32, double-buffered
// async-DMA staging. Wave w computes rows [16w,16w+16) x 128 cols as
// 8 wmma accumulators (A fragment reused 8x per k-step).
// ---------------------------------------------------------------------------
template <int M, int N, int K>
__global__ __launch_bounds__(256) void gemm_bf16_nt(
    const u16* __restrict__ A, const u16* __restrict__ Bw,
    const float* __restrict__ bias, float* __restrict__ C)
{
  __shared__ __attribute__((aligned(16))) u16 As[2][128 * 32];
  __shared__ __attribute__((aligned(16))) u16 Bs[2][128 * 32];
  const int tid  = threadIdx.x;
  const int wave = tid >> 5, lane = tid & 31;
  const int half = lane >> 4, ln = lane & 15;
  const int m0 = blockIdx.y * 128, n0 = blockIdx.x * 128;

  // This thread's two 16B staging chunks within a 128x32 bf16 tile.
  const int r0 = tid >> 2,           c0 = (tid & 3) << 3;
  const int r1 = (tid + 256) >> 2,   c1 = (tid & 3) << 3;

  F8 acc[8] = {};
  constexpr int NK = K / 32;

  auto stage = [&](int t, int buf) {
    const int k0 = t * 32;
    async_copy_b128(&As[buf][r0 * 32 + c0], &A [(size_t)(m0 + r0) * K + k0 + c0]);
    async_copy_b128(&As[buf][r1 * 32 + c1], &A [(size_t)(m0 + r1) * K + k0 + c1]);
    async_copy_b128(&Bs[buf][r0 * 32 + c0], &Bw[(size_t)(n0 + r0) * K + k0 + c0]);
    async_copy_b128(&Bs[buf][r1 * 32 + c1], &Bw[(size_t)(n0 + r1) * K + k0 + c1]);
  };

  stage(0, 0);
  for (int t = 0; t < NK; ++t) {
    const int cur = t & 1;
    if (t + 1 < NK) { stage(t + 1, cur ^ 1); wait_asynccnt<4>(); }
    else            { wait_asynccnt<0>(); }
    __syncthreads();

    // A fragment: lane ln = row, half selects K groups {8h..8h+8, 16+8h..+8}.
    BFrag af;
    const u16* ar = &As[cur][(16 * wave + ln) * 32];
    af.q[0] = *(const uint4*)(ar + 8 * half);
    af.q[1] = *(const uint4*)(ar + 16 + 8 * half);
#pragma unroll
    for (int ct = 0; ct < 8; ++ct) {
      BFrag bf;
      const u16* br = &Bs[cur][(16 * ct + ln) * 32];
      bf.q[0] = *(const uint4*)(br + 8 * half);
      bf.q[1] = *(const uint4*)(br + 16 + 8 * half);
      acc[ct].v = wmma_bf16(af, bf, acc[ct].v);
    }
    __syncthreads();
  }

  // C layout: VGPR v -> row v + 8*half; lane ln -> col.
#pragma unroll
  for (int ct = 0; ct < 8; ++ct) {
    const int col = n0 + 16 * ct + ln;
    const float bv = bias[col];
#pragma unroll
    for (int v = 0; v < 8; ++v) {
      const int row = m0 + 16 * wave + v + 8 * half;
      C[(size_t)row * N + col] = acc[ct].f[v] + bv;
    }
  }
}

// ---------------------------------------------------------------------------
// Per-head LayerNorm + RoPE, fp32 -> bf16, transposed to [B, H, S, Dh].
// Block (32, 8): one wave per head-slot; 4 elements per lane (Dh=128).
// ---------------------------------------------------------------------------
__device__ __forceinline__ void ln_rope_head(
    const float* __restrict__ x, const float* __restrict__ g,
    const float* __restrict__ bta, const float* cs, const float* sn,
    int lane, u16* __restrict__ dst)
{
  float xv[4]; float sum = 0.f;
#pragma unroll
  for (int j = 0; j < 4; ++j) { xv[j] = x[lane + 32 * j]; sum += xv[j]; }
#pragma unroll
  for (int off = 16; off > 0; off >>= 1) sum += __shfl_xor(sum, off, 32);
  const float mu = sum * 0.0078125f;
  float var = 0.f;
#pragma unroll
  for (int j = 0; j < 4; ++j) { float t = xv[j] - mu; var += t * t; }
#pragma unroll
  for (int off = 16; off > 0; off >>= 1) var += __shfl_xor(var, off, 32);
  const float rs = rsqrtf(var * 0.0078125f + 1e-5f);
  float yv[4];
#pragma unroll
  for (int j = 0; j < 4; ++j) {
    const int d = lane + 32 * j;
    yv[j] = (xv[j] - mu) * rs * g[d] + bta[d];
  }
  // rotate_half: d and d+-64 live in the same lane (j and j+-2).
  const float ro[4] = { -yv[2], -yv[3], yv[0], yv[1] };
#pragma unroll
  for (int j = 0; j < 4; ++j)
    dst[lane + 32 * j] = f2bf(yv[j] * cs[j] + ro[j] * sn[j]);
}

__global__ __launch_bounds__(256) void ln_rope_kernel(
    const float* __restrict__ qf, const float* __restrict__ kf,
    const float* __restrict__ vf,
    const float* __restrict__ cosp, const float* __restrict__ sinp,
    const float* __restrict__ qg, const float* __restrict__ qb,
    const float* __restrict__ kg, const float* __restrict__ kb,
    u16* __restrict__ qo, u16* __restrict__ ko, u16* __restrict__ vo)
{
  const int s = blockIdx.x, b = blockIdx.y;
  const int lane = threadIdx.x, slot = threadIdx.y;
  float cs[4], sn[4];
#pragma unroll
  for (int j = 0; j < 4; ++j) {
    cs[j] = cosp[s * 128 + lane + 32 * j];
    sn[j] = sinp[s * 128 + lane + 32 * j];
  }
  for (int h = slot; h < 16; h += 8) {
    const float* x = qf + (((size_t)(b * 1024 + s)) * 16 + h) * 128;
    u16* dst = qo + (((size_t)b * 16 + h) * 1024 + s) * 128;
    ln_rope_head(x, qg, qb, cs, sn, lane, dst);
  }
  if (slot < 4) {
    const float* xk = kf + (((size_t)(b * 1024 + s)) * 4 + slot) * 128;
    u16* dk = ko + (((size_t)b * 4 + slot) * 1024 + s) * 128;
    ln_rope_head(xk, kg, kb, cs, sn, lane, dk);
    const float* xv = vf + (((size_t)(b * 1024 + s)) * 4 + slot) * 128;
    u16* dv = vo + (((size_t)b * 4 + slot) * 1024 + s) * 128;
#pragma unroll
    for (int j = 0; j < 4; ++j)
      dv[lane + 32 * j] = f2bf(xv[lane + 32 * j]);
  }
}

// ---------------------------------------------------------------------------
// Causal flash attention. Grid (S/64, H, B), block 128 (4 waves).
// Q/K tiles staged by async DMA; V is transposed into LDS ([d][key]) with
// register loads, overlapping the in-flight K DMA. Online softmax with
// per-row stats per (half, VGPR-row); 16-lane shfl_xor reductions.
// ---------------------------------------------------------------------------
__global__ __launch_bounds__(128) void attn_kernel(
    const u16* __restrict__ q_bf,  // [B,H,S,128]
    const u16* __restrict__ k_bf,  // [B,Hkv,S,128]
    const u16* __restrict__ v_bf,  // [B,Hkv,S,128]
    u16* __restrict__ o_bf)        // [B,S,H*128]
{
  constexpr int S = 1024, H = 16, HKV = 4, DH = 128;
  __shared__ __attribute__((aligned(16))) u16 Qs[64 * 128];
  __shared__ __attribute__((aligned(16))) u16 Ks[64 * 128];
  __shared__ __attribute__((aligned(16))) u16 Vt[128 * 64];   // [d][key]
  __shared__ __attribute__((aligned(16))) u16 Ps[64 * 64];
  const int bx = blockIdx.x, h = blockIdx.y, b = blockIdx.z;
  const int hkv = h >> 2;
  const int tid = threadIdx.x;
  const int wave = tid >> 5, lane = tid & 31;
  const int half = lane >> 4, ln = lane & 15;

  const u16* Qg = q_bf + (((size_t)b * H + h) * S + bx * 64) * DH;
  const u16* Kg = k_bf + ((size_t)b * HKV + hkv) * S * DH;
  const u16* Vg = v_bf + ((size_t)b * HKV + hkv) * S * DH;

  // Async-stage Q tile once (64x128 bf16 = 1024 x b128, 8 per thread).
#pragma unroll
  for (int j = 0; j < 8; ++j) {
    const int i = tid + 128 * j;
    const int r = i >> 4, c = (i & 15) << 3;
    async_copy_b128(&Qs[r * 128 + c], &Qg[(size_t)r * DH + c]);
  }

  F8 o_acc[8] = {};
  float m_r[8], l_r[8];
#pragma unroll
  for (int v = 0; v < 8; ++v) { m_r[v] = -1e30f; l_r[v] = 0.f; }
  const float scale = 0.08838834764831845f;  // 1/sqrt(128)

  for (int kt = 0; kt <= bx; ++kt) {
    // Async-stage K tile; overlap the V transpose with the DMA.
#pragma unroll
    for (int j = 0; j < 8; ++j) {
      const int i = tid + 128 * j;
      const int r = i >> 4, c = (i & 15) << 3;
      async_copy_b128(&Ks[r * 128 + c], &Kg[(size_t)(kt * 64 + r) * DH + c]);
    }
    for (int i = tid; i < 64 * 64; i += 128) {
      const int r = i >> 6, cu = i & 63;
      const u32 w = *(const u32*)&Vg[(size_t)(kt * 64 + r) * DH + cu * 2];
      Vt[(2 * cu) * 64 + r]     = (u16)(w & 0xffffu);
      Vt[(2 * cu + 1) * 64 + r] = (u16)(w >> 16);
    }
    wait_asynccnt<0>();
    __syncthreads();

    // S = Q K^T : 4 col-tiles x 4 d-chunks of 32.
    F8 sa[4] = {};
#pragma unroll
    for (int kk = 0; kk < 4; ++kk) {
      BFrag af;
      const u16* ar = &Qs[(16 * wave + ln) * 128 + 32 * kk];
      af.q[0] = *(const uint4*)(ar + 8 * half);
      af.q[1] = *(const uint4*)(ar + 16 + 8 * half);
#pragma unroll
      for (int ct = 0; ct < 4; ++ct) {
        BFrag bf;
        const u16* br = &Ks[(16 * ct + ln) * 128 + 32 * kk];
        bf.q[0] = *(const uint4*)(br + 8 * half);
        bf.q[1] = *(const uint4*)(br + 16 + 8 * half);
        sa[ct].v = wmma_bf16(af, bf, sa[ct].v);
      }
    }

    // Scale + causal mask (only the diagonal tile needs masking).
    const bool diag = (kt == bx);
#pragma unroll
    for (int ct = 0; ct < 4; ++ct)
#pragma unroll
      for (int v = 0; v < 8; ++v) {
        float val = sa[ct].f[v] * scale;
        if (diag) {
          const int key = 16 * ct + ln;
          const int qr  = 16 * wave + v + 8 * half;
          if (key > qr) val = -1e30f;
        }
        sa[ct].f[v] = val;
      }

    // Online softmax per row.
#pragma unroll
    for (int v = 0; v < 8; ++v) {
      float mx = sa[0].f[v];
#pragma unroll
      for (int ct = 1; ct < 4; ++ct) mx = fmaxf(mx, sa[ct].f[v]);
#pragma unroll
      for (int off = 8; off > 0; off >>= 1) mx = fmaxf(mx, __shfl_xor(mx, off, 32));
      const float mnew = fmaxf(m_r[v], mx);
      const float alpha = __expf(m_r[v] - mnew);
      float rs = 0.f;
#pragma unroll
      for (int ct = 0; ct < 4; ++ct) {
        const float p = __expf(sa[ct].f[v] - mnew);
        sa[ct].f[v] = p;
        rs += p;
      }
#pragma unroll
      for (int off = 8; off > 0; off >>= 1) rs += __shfl_xor(rs, off, 32);
      l_r[v] = l_r[v] * alpha + rs;
      m_r[v] = mnew;
#pragma unroll
      for (int dt = 0; dt < 8; ++dt) o_acc[dt].f[v] *= alpha;
    }

    // P -> LDS (bf16), wave-private rows.
#pragma unroll
    for (int ct = 0; ct < 4; ++ct)
#pragma unroll
      for (int v = 0; v < 8; ++v)
        Ps[(16 * wave + v + 8 * half) * 64 + 16 * ct + ln] = f2bf(sa[ct].f[v]);
    __syncthreads();

    // O += P V : 8 d-tiles x 2 key-chunks of 32.
#pragma unroll
    for (int kk = 0; kk < 2; ++kk) {
      BFrag af;
      const u16* ar = &Ps[(16 * wave + ln) * 64 + 32 * kk];
      af.q[0] = *(const uint4*)(ar + 8 * half);
      af.q[1] = *(const uint4*)(ar + 16 + 8 * half);
#pragma unroll
      for (int dt = 0; dt < 8; ++dt) {
        BFrag bf;
        const u16* br = &Vt[(16 * dt + ln) * 64 + 32 * kk];
        bf.q[0] = *(const uint4*)(br + 8 * half);
        bf.q[1] = *(const uint4*)(br + 16 + 8 * half);
        o_acc[dt].v = wmma_bf16(af, bf, o_acc[dt].v);
      }
    }
    __syncthreads();
  }

  // Normalize and store bf16 into [B,S,H*Dh] for the output projection.
#pragma unroll
  for (int v = 0; v < 8; ++v) {
    const float inv = 1.f / l_r[v];
    const int row = bx * 64 + 16 * wave + v + 8 * half;
#pragma unroll
    for (int dt = 0; dt < 8; ++dt) {
      const int col = h * 128 + 16 * dt + ln;
      o_bf[((size_t)(b * 1024 + row)) * 2048 + col] = f2bf(o_acc[dt].f[v] * inv);
    }
  }
}

// ---------------------------------------------------------------------------
// Launch
// ---------------------------------------------------------------------------
extern "C" void kernel_launch(void* const* d_in, const int* in_sizes, int n_in,
                              void* d_out, int out_size, void* d_ws, size_t ws_size,
                              hipStream_t stream) {
  (void)in_sizes; (void)n_in; (void)out_size; (void)ws_size;
  const float* hs   = (const float*)d_in[0];
  const float* cosp = (const float*)d_in[1];
  const float* sinp = (const float*)d_in[2];
  const float* Wq   = (const float*)d_in[3];
  const float* bq   = (const float*)d_in[4];
  const float* Wk   = (const float*)d_in[5];
  const float* bk   = (const float*)d_in[6];
  const float* Wv   = (const float*)d_in[7];
  const float* bv   = (const float*)d_in[8];
  const float* Wo   = (const float*)d_in[9];
  const float* bo   = (const float*)d_in[10];
  const float* qg   = (const float*)d_in[11];
  const float* qb   = (const float*)d_in[12];
  const float* kg   = (const float*)d_in[13];
  const float* kb   = (const float*)d_in[14];
  float* out = (float*)d_out;

  constexpr int B = 2, S = 1024, DM = 2048, H = 16, HKV = 4, DH = 128;
  constexpr int M = B * S;              // 2048 rows
  constexpr int NQ = H * DH;            // 2048
  constexpr int NKV = HKV * DH;         // 512

  char* p = (char*)d_ws;
  auto alloc = [&](size_t bytes) {
    char* r = p;
    p += (bytes + 255) & ~(size_t)255;
    return r;
  };
  u16* hs_bf   = (u16*)alloc((size_t)M * DM * 2);
  u16* wq_bf   = (u16*)alloc((size_t)NQ * DM * 2);
  u16* wk_bf   = (u16*)alloc((size_t)NKV * DM * 2);
  u16* wv_bf   = (u16*)alloc((size_t)NKV * DM * 2);
  u16* wo_bf   = (u16*)alloc((size_t)DM * NQ * 2);
  float* qf    = (float*)alloc((size_t)M * NQ * 4);
  float* kf    = (float*)alloc((size_t)M * NKV * 4);
  float* vf    = (float*)alloc((size_t)M * NKV * 4);
  u16* q_bf    = (u16*)alloc((size_t)B * H * S * DH * 2);
  u16* k_bf    = (u16*)alloc((size_t)B * HKV * S * DH * 2);
  u16* v_bf    = (u16*)alloc((size_t)B * HKV * S * DH * 2);
  u16* attn_bf = (u16*)alloc((size_t)M * NQ * 2);

  // 1) bf16 conversions.
  cvt_bf16_kernel<<<(M * DM) / 256, 256, 0, stream>>>(hs, hs_bf, M * DM);
  cvt_bf16_kernel<<<(NQ * DM) / 256, 256, 0, stream>>>(Wq, wq_bf, NQ * DM);
  cvt_bf16_kernel<<<(NKV * DM) / 256, 256, 0, stream>>>(Wk, wk_bf, NKV * DM);
  cvt_bf16_kernel<<<(NKV * DM) / 256, 256, 0, stream>>>(Wv, wv_bf, NKV * DM);
  cvt_bf16_kernel<<<(DM * NQ) / 256, 256, 0, stream>>>(Wo, wo_bf, DM * NQ);

  // 2) QKV projections (WMMA GEMM, fp32 out).
  gemm_bf16_nt<M, NQ, DM><<<dim3(NQ / 128, M / 128), 256, 0, stream>>>(hs_bf, wq_bf, bq, qf);
  gemm_bf16_nt<M, NKV, DM><<<dim3(NKV / 128, M / 128), 256, 0, stream>>>(hs_bf, wk_bf, bk, kf);
  gemm_bf16_nt<M, NKV, DM><<<dim3(NKV / 128, M / 128), 256, 0, stream>>>(hs_bf, wv_bf, bv, vf);

  // 3) LayerNorm + RoPE + transpose to [B,H,S,Dh] bf16.
  ln_rope_kernel<<<dim3(S, B), dim3(32, 8), 0, stream>>>(
      qf, kf, vf, cosp, sinp, qg, qb, kg, kb, q_bf, k_bf, v_bf);

  // 4) Causal flash attention.
  attn_kernel<<<dim3(S / 64, H, B), 128, 0, stream>>>(q_bf, k_bf, v_bf, attn_bf);

  // 5) Output projection straight into d_out (fp32).
  gemm_bf16_nt<M, NQ, DM><<<dim3(NQ / 128, M / 128), 256, 0, stream>>>(attn_bf, wo_bf, bo, out);
}
